// MultiHeadAttention_81501299408929
// MI455X (gfx1250) — matrix operand order repaired and checked by reading, not compiled, not run on hardware
//
#include <hip/hip_runtime.h>
#include <hip/hip_bf16.h>
#include <math.h>
#include <stdint.h>

typedef __bf16 bf16;
typedef __attribute__((ext_vector_type(16))) __bf16 v16bf;
typedef __attribute__((ext_vector_type(8)))  float  v8f;
typedef __attribute__((ext_vector_type(4)))  int    v4i;
typedef __attribute__((ext_vector_type(4)))  unsigned v4u;
typedef __attribute__((ext_vector_type(8)))  int    v8i;

#define B_   4
#define S_   2048
#define D_   1024
#define H_   16
#define HD_  64
#define MTOT (B_ * S_)   // 8192

// GEMM tiling
#define GTM 128
#define GTN 64
#define GTK 32

// Attention tiling
#define AQ   128   // q rows per workgroup (16 per wave, 8 waves)
#define AKV  64    // keys per block
#define ASCALE 0.125f           // 1/sqrt(64)
#define LOG2E  1.4426950408889634f

#define AS1 __attribute__((address_space(1)))
#define AS3 __attribute__((address_space(3)))

#define HAS_TDM (__has_builtin(__builtin_amdgcn_tensor_load_to_lds) && \
                 __has_builtin(__builtin_amdgcn_s_wait_tensorcnt))

static __device__ __forceinline__ v8f wmma_bf16(v16bf a, v16bf b, v8f c) {
  // (neg_a, A, neg_b, B, c_mod, C, reuse_a, reuse_b)
  return __builtin_amdgcn_wmma_f32_16x16x32_bf16(false, a, false, b, (short)0, c,
                                                 false, false);
}

// 16-byte global -> LDS copy on the CDNA5 async path (ASYNCcnt) when available.
static __device__ __forceinline__ void async_copy_b128(const void* g, void* l) {
#if __has_builtin(__builtin_amdgcn_global_load_async_to_lds_b128)
  __builtin_amdgcn_global_load_async_to_lds_b128(
      (AS1 v4i*)(uintptr_t)g, (AS3 v4i*)(unsigned)(uintptr_t)l, 0, 0);
#else
  *(uint4*)l = *(const uint4*)g;
#endif
}

static __device__ __forceinline__ void async_wait0() {
#if __has_builtin(__builtin_amdgcn_s_wait_asynccnt)
  __builtin_amdgcn_s_wait_asynccnt(0);
#else
  asm volatile("s_wait_asynccnt 0" ::: "memory");
#endif
}

#if HAS_TDM
// TDM: DMA one 64-row x 64-elem (bf16) tile, row stride 1024 elems, into LDS.
// D# layout per CDNA5 ISA (2-D tensor, groups 2/3 zero).
static __device__ __forceinline__ void tdm_load_tile_64x64(const bf16* g, bf16* l) {
  unsigned long long ga = (unsigned long long)(uintptr_t)g;   // 57-bit global addr
  unsigned lds = (unsigned)(uintptr_t)l;                      // LDS byte offset
  v4u g0;
  g0[0] = 1u;                                        // count=1, user D#
  g0[1] = lds;                                       // lds_addr [63:32]
  g0[2] = (unsigned)(ga & 0xFFFFFFFFu);              // global_addr[31:0]
  g0[3] = (unsigned)((ga >> 32) & 0x1FFFFFFu)        // global_addr[56:32]
          | (2u << 30);                              // type=2 ("image")
  v8i g1;
  g1[0] = (int)(1u << 16);        // workgroup_mask=0, data_size=1 (2 bytes)
  g1[1] = (int)(64u << 16);       // tensor_dim0 = 64  (bits 79:48, low 16)
  g1[2] = (int)(2048u << 16);     // tensor_dim0 hi=0; tensor_dim1 = 2048 (low 16)
  g1[3] = (int)(64u << 16);       // tensor_dim1 hi=0; tile_dim0 = 64
  g1[4] = 64;                     // tile_dim1 = 64; tile_dim2 = 0
  g1[5] = 1024;                   // tensor_dim0_stride = 1024 elems
  g1[6] = 0;                      // stride hi; tensor_dim1_stride lo = 0
  g1[7] = 0;
  v4i zz4 = {0, 0, 0, 0};
  v8i zz8 = {0, 0, 0, 0, 0, 0, 0, 0};
  // 6-arg toolchain form: (g0, g1, g2, g3, extra, cpol)
  __builtin_amdgcn_tensor_load_to_lds(g0, g1, zz4, zz4, zz8, 0);
}
#endif

union AFrag { v16bf v; unsigned u[8]; };

// ---------------- fp32 -> bf16 conversion (4 elems / thread) ----------------
__global__ __launch_bounds__(256)
void cvt_f32_bf16(const float* __restrict__ src, bf16* __restrict__ dst, long n) {
  long i = ((long)blockIdx.x * blockDim.x + threadIdx.x) * 4;
  if (i + 3 < n) {
    float4 f = *(const float4*)(src + i);
    union { bf16 h[4]; unsigned long long u; } o;
    o.h[0] = (bf16)f.x; o.h[1] = (bf16)f.y; o.h[2] = (bf16)f.z; o.h[3] = (bf16)f.w;
    *(unsigned long long*)(dst + i) = o.u;
  }
}

// ---------------- bf16 GEMM: C[M,N] = A[M,K] @ W[K,N] (+bias) ----------------
// Block tile 128x64, K-step 32. 8 waves, each wave a 32x32 tile (2x2 wmma).
// A staged [row][k], B staged [k][n]; both via linear async B128 copies.
template <bool WRITE_F32>
__global__ __launch_bounds__(256)
void gemm_bf16(const bf16* __restrict__ A, const bf16* __restrict__ Bw,
               bf16* __restrict__ Cb, float* __restrict__ Cf,
               const float* __restrict__ bias, int M, int N, int K) {
  __shared__ __align__(16) bf16 As[GTM * GTK];   // [row][k]  8 KB
  __shared__ __align__(16) bf16 Bs[GTK * GTN];   // [k][n]    4 KB

  const int tid  = threadIdx.x;
  const int wid  = tid >> 5;
  const int lane = tid & 31;
  const int half = (lane >> 4) & 1;
  const int ln   = lane & 15;
  const int m0 = blockIdx.y * GTM;
  const int n0 = blockIdx.x * GTN;
  const int wm = (wid >> 1) * 32;   // wave M offset: 0,32,64,96
  const int wn = (wid & 1) * 32;    // wave N offset: 0,32

  const v8f vz = {0.f, 0.f, 0.f, 0.f, 0.f, 0.f, 0.f, 0.f};
  v8f acc[2][2];
  acc[0][0] = vz; acc[0][1] = vz; acc[1][0] = vz; acc[1][1] = vz;

  for (int kk = 0; kk < K; kk += GTK) {
    __syncthreads();
    // A tile 128x32 = 512 B128 chunks (2/thread), async to LDS
    #pragma unroll
    for (int j = 0; j < 2; ++j) {
      int l   = tid * 2 + j;        // 0..511
      int row = l >> 2;             // 0..127
      int kq  = (l & 3) * 8;        // 0,8,16,24
      async_copy_b128(&A[(long)(m0 + row) * K + kk + kq], &As[row * GTK + kq]);
    }
    // B tile 32x64 = 256 B128 chunks (1/thread), async to LDS
    {
      int kr = tid >> 3;            // 0..31
      int np = (tid & 7) * 8;       // 0..56
      async_copy_b128(&Bw[(long)(kk + kr) * N + n0 + np], &Bs[kr * GTN + np]);
    }
    // Prefetch next K-step while this one is consumed
    if (kk + GTK < K) {
      int row = tid >> 1; int kq = (tid & 1) * 16;
      __builtin_prefetch(&A[(long)(m0 + row) * K + kk + GTK + kq], 0, 0);
      if (tid < 64) {
        int kr = tid >> 1; int np = (tid & 1) * 32;
        __builtin_prefetch(&Bw[(long)(kk + GTK + kr) * N + n0 + np], 0, 0);
      }
    }
    async_wait0();
    __syncthreads();

    // A frags: lane ln = row, element pair (v): k = (v/4)*16 + half*8 + (v&3)*2
    AFrag af[2], bfg[2];
    #pragma unroll
    for (int i = 0; i < 2; ++i) {
      int mo = wm + i * 16;
      #pragma unroll
      for (int v = 0; v < 8; ++v) {
        int k = ((v >> 2) * 16) + half * 8 + (v & 3) * 2;
        af[i].u[v] = *(const unsigned*)&As[(mo + ln) * GTK + k];
      }
    }
    // B frags from [k][n]: element pair (v): k = (v/4)*16 + half*8 + (v&3)*2, rows k and k+1
    #pragma unroll
    for (int j = 0; j < 2; ++j) {
      int no = wn + j * 16;
      #pragma unroll
      for (int v = 0; v < 8; ++v) {
        int k = ((v >> 2) * 16) + half * 8 + (v & 3) * 2;
        unsigned lo = *(const unsigned short*)&Bs[k * GTN + no + ln];
        unsigned hi = *(const unsigned short*)&Bs[(k + 1) * GTN + no + ln];
        bfg[j].u[v] = lo | (hi << 16);
      }
    }
    #pragma unroll
    for (int i = 0; i < 2; ++i)
      #pragma unroll
      for (int j = 0; j < 2; ++j)
        acc[i][j] = wmma_bf16(af[i].v, bfg[j].v, acc[i][j]);
  }

  // Epilogue: D layout -> lane ln = col, M = half*8 + v
  #pragma unroll
  for (int i = 0; i < 2; ++i) {
    #pragma unroll
    for (int j = 0; j < 2; ++j) {
      #pragma unroll
      for (int v = 0; v < 8; ++v) {
        int row = m0 + wm + i * 16 + half * 8 + v;
        int col = n0 + wn + j * 16 + ln;
        float val = acc[i][j][v];
        if (WRITE_F32) Cf[(long)row * N + col] = val + bias[col];
        else           Cb[(long)row * N + col] = (bf16)val;
      }
    }
  }
}

// ---------------- Flash attention: causal, per (b, h, 128 q rows) ----------------
__global__ __launch_bounds__(256)
void attn_kernel(const bf16* __restrict__ Q, const bf16* __restrict__ Kx,
                 const bf16* __restrict__ V, bf16* __restrict__ O) {
  __shared__ __align__(16) bf16 Ks[AKV * HD_];     // [key][dim]   8 KB
  __shared__ __align__(16) bf16 Vs[AKV * HD_];     // [key][dim]   8 KB
  __shared__ __align__(16) bf16 Pb[8 * 16 * 64];   // wave-private P  16 KB

  const int tid  = threadIdx.x;
  const int wid  = tid >> 5;
  const int lane = tid & 31;
  const int half = (lane >> 4) & 1;
  const int ln   = lane & 15;

  const int qb = blockIdx.x;
  const int h  = blockIdx.y;
  const int b  = blockIdx.z;

  const int qg0  = qb * AQ;
  const int qw   = qg0 + wid * 16;            // this wave's q base
  const long base = (long)b * S_ * D_ + h * HD_;

  // Q as two A-frags (dim 0..31, 32..63), kept in registers for the whole pass
  AFrag qa[2];
  #pragma unroll
  for (int t = 0; t < 2; ++t)
    #pragma unroll
    for (int v = 0; v < 8; ++v) {
      int kd = t * 32 + ((v >> 2) * 16) + half * 8 + (v & 3) * 2;
      qa[t].u[v] = *(const unsigned*)&Q[base + (long)(qw + ln) * D_ + kd];
    }

  const v8f vz = {0.f, 0.f, 0.f, 0.f, 0.f, 0.f, 0.f, 0.f};
  v8f acc[4]; acc[0] = vz; acc[1] = vz; acc[2] = vz; acc[3] = vz;
  float mrow[8], lrow[8];
  #pragma unroll
  for (int v = 0; v < 8; ++v) { mrow[v] = -INFINITY; lrow[v] = 0.f; }

  bf16* pp = &Pb[wid * 16 * 64];

  const int nkb = (qg0 + AQ) / AKV;   // key blocks needed for this q block
  for (int kb = 0; kb < nkb; ++kb) {
    const int kbase = kb * AKV;
    __syncthreads();
#if HAS_TDM
    // One TDM descriptor per tile: 64 rows x 128B, row stride 2048B. Wave 0
    // issues both DMAs and drains TENSORcnt; the barrier publishes to the WG.
    if (wid == 0) {
      tdm_load_tile_64x64(&Kx[base + (long)kbase * D_], Ks);
      tdm_load_tile_64x64(&V [base + (long)kbase * D_], Vs);
    }
#else
    // K and V tiles, linear [key][dim], 512 B128 chunks each (2+2 per thread)
    #pragma unroll
    for (int j = 0; j < 2; ++j) {
      int l   = tid * 2 + j;      // 0..511
      int key = l >> 3;           // 0..63
      int dof = (l & 7) * 8;      // 0..56
      async_copy_b128(&Kx[base + (long)(kbase + key) * D_ + dof], &Ks[key * HD_ + dof]);
      async_copy_b128(&V [base + (long)(kbase + key) * D_ + dof], &Vs[key * HD_ + dof]);
    }
#endif
    // Prefetch next key block while this one is consumed
    if (kb + 1 < nkb) {
      int key = tid >> 2; int dof = (tid & 3) * 16;
      __builtin_prefetch(&Kx[base + (long)(kbase + AKV + key) * D_ + dof], 0, 0);
      __builtin_prefetch(&V [base + (long)(kbase + AKV + key) * D_ + dof], 0, 0);
    }
#if HAS_TDM
    if (wid == 0) __builtin_amdgcn_s_wait_tensorcnt(0);
#else
    async_wait0();
#endif
    __syncthreads();

    if (kbase <= qw + 15) {   // wave-uniform: skip fully-masked key blocks
      // S = Q @ K^T : B-frag straight from Ks[key][dim] ([n][k] layout)
      v8f sc[4];
      #pragma unroll
      for (int nt = 0; nt < 4; ++nt) {
        AFrag b0, b1;
        #pragma unroll
        for (int v = 0; v < 8; ++v) {
          int kd = ((v >> 2) * 16) + half * 8 + (v & 3) * 2;
          b0.u[v] = *(const unsigned*)&Ks[(nt * 16 + ln) * HD_ + kd];
          b1.u[v] = *(const unsigned*)&Ks[(nt * 16 + ln) * HD_ + 32 + kd];
        }
        v8f c = vz;
        c = wmma_bf16(qa[0].v, b0.v, c);
        c = wmma_bf16(qa[1].v, b1.v, c);
        sc[nt] = c;
      }
      // Online softmax (row = half*8+v spans the 16 lanes of a half-wave)
      float pv[4][8];
      #pragma unroll
      for (int v = 0; v < 8; ++v) {
        int row = qw + half * 8 + v;
        float rmax = mrow[v];
        #pragma unroll
        for (int nt = 0; nt < 4; ++nt) {
          float x = sc[nt][v] * ASCALE;
          int col = kbase + nt * 16 + ln;
          x = (col <= row) ? x : -INFINITY;
          pv[nt][v] = x;
          rmax = fmaxf(rmax, x);
        }
        rmax = fmaxf(rmax, __shfl_xor(rmax, 1, 32));
        rmax = fmaxf(rmax, __shfl_xor(rmax, 2, 32));
        rmax = fmaxf(rmax, __shfl_xor(rmax, 4, 32));
        rmax = fmaxf(rmax, __shfl_xor(rmax, 8, 32));
        float corr = exp2f((mrow[v] - rmax) * LOG2E);
        float ls = 0.f;
        #pragma unroll
        for (int nt = 0; nt < 4; ++nt) {
          float p = exp2f((pv[nt][v] - rmax) * LOG2E);
          pv[nt][v] = p; ls += p;
        }
        ls += __shfl_xor(ls, 1, 32);
        ls += __shfl_xor(ls, 2, 32);
        ls += __shfl_xor(ls, 4, 32);
        ls += __shfl_xor(ls, 8, 32);
        lrow[v] = lrow[v] * corr + ls;
        mrow[v] = rmax;
        #pragma unroll
        for (int nt = 0; nt < 4; ++nt) acc[nt][v] *= corr;
      }
      // C-layout -> A-frag layout via wave-private LDS (in-order within a wave)
      #pragma unroll
      for (int nt = 0; nt < 4; ++nt)
        #pragma unroll
        for (int v = 0; v < 8; ++v)
          pp[(half * 8 + v) * 64 + nt * 16 + ln] = (bf16)pv[nt][v];

      AFrag pa[2];
      #pragma unroll
      for (int t = 0; t < 2; ++t)
        #pragma unroll
        for (int v = 0; v < 8; ++v) {
          int key = t * 32 + ((v >> 2) * 16) + half * 8 + (v & 3) * 2;
          pa[t].u[v] = *(const unsigned*)&pp[ln * 64 + key];
        }
      // O += P @ V : B-frag from Vs[key][dim] with paired 16-bit loads
      #pragma unroll
      for (int nt = 0; nt < 4; ++nt) {
        AFrag v0, v1;
        #pragma unroll
        for (int v = 0; v < 8; ++v) {
          int ky0 = ((v >> 2) * 16) + half * 8 + (v & 3) * 2;        // koff 0
          int ky1 = 32 + ky0;                 // koff 32
          int dim = nt * 16 + ln;
          unsigned lo0 = *(const unsigned short*)&Vs[ky0 * HD_ + dim];
          unsigned hi0 = *(const unsigned short*)&Vs[(ky0 + 1) * HD_ + dim];
          unsigned lo1 = *(const unsigned short*)&Vs[ky1 * HD_ + dim];
          unsigned hi1 = *(const unsigned short*)&Vs[(ky1 + 1) * HD_ + dim];
          v0.u[v] = lo0 | (hi0 << 16);
          v1.u[v] = lo1 | (hi1 << 16);
        }
        acc[nt] = wmma_bf16(pa[0].v, v0.v, acc[nt]);
        acc[nt] = wmma_bf16(pa[1].v, v1.v, acc[nt]);
      }
    }
  }

  // Normalize and store O (bf16) for the output projection
  #pragma unroll
  for (int v = 0; v < 8; ++v) {
    float inv = 1.0f / lrow[v];
    int row = qw + half * 8 + v;
    #pragma unroll
    for (int nt = 0; nt < 4; ++nt)
      O[base + (long)row * D_ + nt * 16 + ln] = (bf16)(acc[nt][v] * inv);
  }
}

// ---------------- launcher ----------------
extern "C" void kernel_launch(void* const* d_in, const int* in_sizes, int n_in,
                              void* d_out, int out_size, void* d_ws, size_t ws_size,
                              hipStream_t stream) {
  (void)in_sizes; (void)n_in; (void)out_size; (void)ws_size;
  const float* x  = (const float*)d_in[0];
  const float* wq = (const float*)d_in[1];
  const float* wk = (const float*)d_in[2];
  const float* wv = (const float*)d_in[3];
  const float* wo = (const float*)d_in[4];
  const float* ob = (const float*)d_in[5];
  float* out = (float*)d_out;

  char* ws = (char*)d_ws;
  const size_t MB = 1ull << 20;
  bf16* Xb  = (bf16*)(ws + 0);        // 16 MB
  bf16* Qb  = (bf16*)(ws + 16 * MB);  // 16 MB
  bf16* Kb  = (bf16*)(ws + 32 * MB);  // 16 MB
  bf16* Vb  = (bf16*)(ws + 48 * MB);  // 16 MB
  bf16* Ob  = (bf16*)(ws + 64 * MB);  // 16 MB
  bf16* Wqb = (bf16*)(ws + 80 * MB);  // 2 MB
  bf16* Wkb = (bf16*)(ws + 82 * MB);
  bf16* Wvb = (bf16*)(ws + 84 * MB);
  bf16* Wob = (bf16*)(ws + 86 * MB);  // end: 88 MB

  const long nX = (long)MTOT * D_;    // 8Mi
  const long nW = (long)D_ * D_;      // 1Mi
  const int thr = 256;
  cvt_f32_bf16<<<(int)(nX / 4 / thr), thr, 0, stream>>>(x,  Xb,  nX);
  cvt_f32_bf16<<<(int)(nW / 4 / thr), thr, 0, stream>>>(wq, Wqb, nW);
  cvt_f32_bf16<<<(int)(nW / 4 / thr), thr, 0, stream>>>(wk, Wkb, nW);
  cvt_f32_bf16<<<(int)(nW / 4 / thr), thr, 0, stream>>>(wv, Wvb, nW);
  cvt_f32_bf16<<<(int)(nW / 4 / thr), thr, 0, stream>>>(wo, Wob, nW);

  dim3 gg(D_ / GTN, MTOT / GTM);      // (16, 64)
  gemm_bf16<false><<<gg, 256, 0, stream>>>(Xb, Wqb, Qb, nullptr, nullptr, MTOT, D_, D_);
  gemm_bf16<false><<<gg, 256, 0, stream>>>(Xb, Wkb, Kb, nullptr, nullptr, MTOT, D_, D_);
  gemm_bf16<false><<<gg, 256, 0, stream>>>(Xb, Wvb, Vb, nullptr, nullptr, MTOT, D_, D_);

  dim3 ga(S_ / AQ, H_, B_);           // (16, 16, 4)
  attn_kernel<<<ga, 256, 0, stream>>>(Qb, Kb, Vb, Ob);

  gemm_bf16<true><<<gg, 256, 0, stream>>>(Ob, Wob, nullptr, out, ob, MTOT, D_, D_);
}
